// MSDeformAttn_SingleLevel_61074434949217
// MI455X (gfx1250) — matrix operand order repaired and checked by reading, not compiled, not run on hardware
//
#include <hip/hip_runtime.h>
#include <hip/hip_bf16.h>
#include <math.h>

// ---------------- problem constants ----------------
#define N_B     4
#define LQ      4096
#define C_DIM   256
#define HEADS   8
#define POINTS  4
#define CPBD    64
#define HW      4096
#define HH      64
#define WW      64
#define DH      32
#define NLQ     (N_B * LQ)                       // 16384 rows for the big GEMMs
#define ROWSCPB (N_B * HEADS * LQ * POINTS)      // 524288 CPB rows
#define KBIG    256                              // K of all generic GEMMs

// ---------------- WMMA types (gfx1250, wave32) ----------------
typedef __attribute__((ext_vector_type(16))) __bf16 v16bf;
typedef __attribute__((ext_vector_type(8)))  __bf16 v8bf;
typedef __attribute__((ext_vector_type(8)))  float  v8f;
typedef __attribute__((ext_vector_type(4)))  int    vec4i;

// CDNA5 async global->LDS DMA (ASYNCcnt-tracked), with safe fallback.
// Probe result (round 2 stderr): arg0 is `int __vector(4)` pointer in the
// global (__device__/AS1) address space; LDS dest is AS3.
#if __has_builtin(__builtin_amdgcn_global_load_async_to_lds_b128)
#define HAVE_ASYNC_LDS 1
typedef __attribute__((address_space(1))) vec4i gv4i;
typedef __attribute__((address_space(3))) vec4i lv4i;
#else
#define HAVE_ASYNC_LDS 0
#endif

__device__ __forceinline__ void wait_async_lds() {
#if HAVE_ASYNC_LDS
#if __has_builtin(__builtin_amdgcn_s_wait_asynccnt)
    __builtin_amdgcn_s_wait_asynccnt(0);
#else
    asm volatile("s_wait_asynccnt 0x0" ::: "memory");
#endif
#endif
}

// A-fragment for v_wmma_f32_16x16x32_bf16 from a row-major bf16 row.
// Per ISA 16-bit A 16x32 layout: lane (m = lane&15, half = lane>>4) holds
// K = {kb*32 + 8*half + 0..7} in v[0..3] and K = {kb*32 + 16 + 8*half + 0..7}
// in v[4..7] -> two contiguous 16-byte chunks.
__device__ __forceinline__ v16bf load_a_frag(const __bf16* __restrict__ Arow,
                                             int kb, int half) {
    const __bf16* p0 = Arow + kb * 32 + 8 * half;
    v8bf lo = *(const v8bf*)p0;
    v8bf hi = *(const v8bf*)(p0 + 16);
    v16bf a;
#pragma unroll
    for (int i = 0; i < 8; ++i) { a[i] = lo[i]; a[8 + i] = hi[i]; }
    return a;
}

// ---------------- f32 -> bf16 convert ----------------
__global__ void f32_to_bf16_kernel(const float* __restrict__ src,
                                   __bf16* __restrict__ dst, int n) {
    int i = blockIdx.x * blockDim.x + threadIdx.x;
    if (i < n) dst[i] = (__bf16)src[i];
}

// ---------------- pack row-major (K x NcAct) weight into B fragments ----------
// Bp[((kb*NT + nt)*32 + lane)*16 + e] = B[kb*32 + 8*(lane>>4) + (e&7) + 16*(e>>3)][nt*16 + (lane&15)]
__global__ void pack_b_kernel(const float* __restrict__ B, __bf16* __restrict__ Bp,
                              int K, int NcAct, int NcPad) {
    int NT = NcPad >> 4;
    int kbCount = K >> 5;
    int total = kbCount * NT * 512;
    int idx = blockIdx.x * blockDim.x + threadIdx.x;
    if (idx >= total) return;
    int e    = idx & 15;
    int lane = (idx >> 4) & 31;
    int t    = idx >> 9;
    int nt   = t % NT;
    int kb   = t / NT;
    int k    = kb * 32 + 8 * (lane >> 4) + (e & 7) + 16 * (e >> 3);
    int col  = nt * 16 + (lane & 15);
    float v = (col < NcAct) ? B[(size_t)k * NcAct + col] : 0.0f;
    Bp[idx] = (__bf16)v;
}

// ---------------- generic WMMA GEMM with async-LDS A staging -----------------
// One wave = one 16x16 C tile; the block's A panels are DMA'd to LDS once and
// shared by all 8 waves. NT (tiles along Nc) is a compile-time constant so the
// panel count PANELS = ceil(8/NT) is static (<= 4 panels = 32 KB LDS).
// EPI 0: row-major f32 store (ldo). EPI 1: value scatter (n*8+h, m, d).
template <int EPI, int NT>
__global__ void wmma_gemm_kernel(const __bf16* __restrict__ A,
                                 const __bf16* __restrict__ Bp,
                                 const float* __restrict__ bias,
                                 float* __restrict__ outF,
                                 int M, int ldo) {
    constexpr int PANELS = (8 + NT - 1) / NT;        // M-tiles touched by block
    constexpr int KB = KBIG / 32;                    // 8 k-blocks
    __shared__ __bf16 smA[PANELS * 16 * KBIG];       // PANELS * 8 KB

    const int lane = threadIdx.x;
    const int half = lane >> 4;
    const int nidx = lane & 15;
    const int tile = blockIdx.x * blockDim.y + threadIdx.y;
    const int mt   = tile / NT;
    const int nt   = tile - mt * NT;
    const int mt0  = (blockIdx.x * 8) / NT;          // first M-tile of block
    const int tid  = threadIdx.y * 32 + lane;

    // ---- stage A panels (16 rows x 256 k each) into LDS ----
    // 16-byte chunks: 512 per panel; chunk c -> panel, row (0..15), 8-elem col.
    for (int c = tid; c < PANELS * 512; c += 256) {
        int panel = c >> 9;
        int rem   = c & 511;
        int row16 = rem >> 5;
        int cc    = rem & 31;
        const __bf16* g = A + ((size_t)((mt0 + panel) * 16 + row16)) * KBIG + cc * 8;
        __bf16* l = smA + (size_t)c * 8;
#if HAVE_ASYNC_LDS
        __builtin_amdgcn_global_load_async_to_lds_b128((gv4i*)g, (lv4i*)l, 0, 0);
#else
        *(uint4*)l = *(const uint4*)g;
#endif
    }
    wait_async_lds();
    __syncthreads();

    if (mt * 16 >= M) return;
    const int pi = mt - mt0;                          // this wave's panel
    const __bf16* arow  = smA + ((size_t)pi * 16 + nidx) * KBIG;
    const __bf16* Bbase = Bp + ((size_t)nt * 32 + lane) * 16;

    v8f acc = {};
#pragma unroll
    for (int kb = 0; kb < KB; ++kb) {
        if (kb + 1 < KB)
            __builtin_prefetch(Bbase + (size_t)(kb + 1) * NT * 512, 0, 3);
        v16bf a = load_a_frag(arow, kb, half);        // ds_load_b128 x2
        v16bf b = *(const v16bf*)(Bbase + (size_t)kb * NT * 512);
        acc = __builtin_amdgcn_wmma_f32_16x16x32_bf16(
                  false, a, false, b, (short)0, acc, false, false);
    }

    const int col = nt * 16 + nidx;
    const float bv = bias ? bias[col] : 0.0f;
#pragma unroll
    for (int r = 0; r < 8; ++r) {
        int m = mt * 16 + r + 8 * half;   // C/D layout: VGPR r -> M = r (+8 hi half)
        float v = acc[r] + bv;
        if (EPI == 0) {
            outF[(size_t)m * ldo + col] = v;
        } else {
            int n = m >> 12, mloc = m & 4095;      // row = n*4096 + mloc
            int h = col >> 5, d = col & 31;        // col = h*32 + d
            outF[((size_t)(n * HEADS + h) * HW + mloc) * DH + d] = v;
        }
    }
}

// ---------------- sampling locations (tanh offsets -> unclipped loc) ----------
__global__ void loc_kernel(const float* __restrict__ offs_raw,
                           const float* __restrict__ ref,
                           float* __restrict__ loc) {
    int idx = blockIdx.x * blockDim.x + threadIdx.x;   // ((n*LQ+lq)*HEADS+h)*POINTS+p
    if (idx >= NLQ * HEADS * POINTS) return;
    int p = idx & 3;
    int t = idx >> 2;
    int h = t & 7;
    int nl = t >> 3;                                   // n*LQ + lq
    const float* o = offs_raw + (size_t)nl * 64 + (h * POINTS + p) * 2;
    float ox = 0.1f * tanhf(o[0]);
    float oy = 0.1f * tanhf(o[1]);
    float lx = 2.0f * ref[nl * 2 + 0] - 1.0f + ox / (float)WW;
    float ly = 2.0f * ref[nl * 2 + 1] - 1.0f + oy / (float)HH;
    loc[(size_t)idx * 2 + 0] = lx;
    loc[(size_t)idx * 2 + 1] = ly;
}

// ---------------- fused CPB MLP: layer1 VALU -> LDS -> WMMA layer2/3 ----------
__global__ void cpb_kernel(const float* __restrict__ loc,
                           const float* __restrict__ ref,
                           const float* __restrict__ W1, const float* __restrict__ b1,
                           const __bf16* __restrict__ W2p, const float* __restrict__ b2,
                           const __bf16* __restrict__ W3p, const float* __restrict__ b3,
                           float* __restrict__ bias8) {
    __shared__ float lds[8 * 16 * 64];
    float* sm = lds + (size_t)threadIdx.y * 16 * 64;
    const int lane = threadIdx.x;
    const int half = lane >> 4;
    const int nidx = lane & 15;
    const int wave = blockIdx.x * blockDim.y + threadIdx.y;
    const int r0 = wave * 16;

    // ---- layer 1: h1 = relu(rel @ W1 + b1), 2 -> 64, into LDS ----
    for (int i = 0; i < 16; ++i) {
        int row = r0 + i;
        int p  = row & 3;
        int t  = row >> 2;
        int lq = t & 4095;
        int t2 = t >> 12;
        int h  = t2 & 7;
        int n  = t2 >> 3;
        int nl = n * LQ + lq;
        int li = ((nl * HEADS + h) * POINTS + p) * 2;
        float rx = loc[li]     - ref[nl * 2 + 0];
        float ry = loc[li + 1] - ref[nl * 2 + 1];
        rx = copysignf(log1pf(fabsf(rx)), rx);
        ry = copysignf(log1pf(fabsf(ry)), ry);
#pragma unroll
        for (int cc = 0; cc < 2; ++cc) {
            int c = lane + 32 * cc;
            float v = rx * W1[c] + ry * W1[64 + c] + b1[c];
            sm[i * 64 + c] = fmaxf(v, 0.0f);
        }
    }
    __syncthreads();

    // ---- layer-2 A fragments (K = 64 -> 2 k-blocks) ----
    v16bf a0, a1;
#pragma unroll
    for (int e = 0; e < 8; ++e) {
        a0[e]     = (__bf16)sm[nidx * 64 + 8 * half + e];
        a0[8 + e] = (__bf16)sm[nidx * 64 + 16 + 8 * half + e];
        a1[e]     = (__bf16)sm[nidx * 64 + 32 + 8 * half + e];
        a1[8 + e] = (__bf16)sm[nidx * 64 + 32 + 16 + 8 * half + e];
    }
    v8f acc[4] = {};
#pragma unroll
    for (int nt = 0; nt < 4; ++nt) {
        v16bf bL = *(const v16bf*)(W2p + ((size_t)(0 * 4 + nt) * 32 + lane) * 16);
        acc[nt] = __builtin_amdgcn_wmma_f32_16x16x32_bf16(
                      false, a0, false, bL, (short)0, acc[nt], false, false);
        v16bf bH = *(const v16bf*)(W2p + ((size_t)(1 * 4 + nt) * 32 + lane) * 16);
        acc[nt] = __builtin_amdgcn_wmma_f32_16x16x32_bf16(
                      false, a1, false, bH, (short)0, acc[nt], false, false);
    }
    __syncthreads();

    // ---- h2 = relu(. + b2) back to LDS (C/D layout -> row major) ----
#pragma unroll
    for (int nt = 0; nt < 4; ++nt) {
        int col = nt * 16 + nidx;
        float bb = b2[col];
#pragma unroll
        for (int r = 0; r < 8; ++r) {
            int m = r + 8 * half;
            sm[m * 64 + col] = fmaxf(acc[nt][r] + bb, 0.0f);
        }
    }
    __syncthreads();

    // ---- layer 3: 64 -> 8 (W3 zero-padded to 16 cols) ----
    v16bf c0, c1;
#pragma unroll
    for (int e = 0; e < 8; ++e) {
        c0[e]     = (__bf16)sm[nidx * 64 + 8 * half + e];
        c0[8 + e] = (__bf16)sm[nidx * 64 + 16 + 8 * half + e];
        c1[e]     = (__bf16)sm[nidx * 64 + 32 + 8 * half + e];
        c1[8 + e] = (__bf16)sm[nidx * 64 + 32 + 16 + 8 * half + e];
    }
    v8f acc3 = {};
    v16bf w3a = *(const v16bf*)(W3p + ((size_t)0 * 32 + lane) * 16);
    acc3 = __builtin_amdgcn_wmma_f32_16x16x32_bf16(
               false, c0, false, w3a, (short)0, acc3, false, false);
    v16bf w3b = *(const v16bf*)(W3p + ((size_t)1 * 32 + lane) * 16);
    acc3 = __builtin_amdgcn_wmma_f32_16x16x32_bf16(
               false, c1, false, w3b, (short)0, acc3, false, false);
    if (nidx < 8) {
        float bb = b3[nidx];
#pragma unroll
        for (int r = 0; r < 8; ++r) {
            int m = r + 8 * half;
            bias8[(size_t)(r0 + m) * 8 + nidx] = acc3[r] + bb;
        }
    }
}

// ---------------- double softmax + tanh(mean CPB bias) -> attn ----------------
__global__ void attn_kernel(const float* __restrict__ aw_raw,
                            const float* __restrict__ bias8,
                            float* __restrict__ attn,
                            float* __restrict__ attn_out) {
    int idx = blockIdx.x * blockDim.x + threadIdx.x;  // (n*LQ+lq)*HEADS + h
    if (idx >= NLQ * HEADS) return;
    int h  = idx & 7;
    int nl = idx >> 3;
    int n  = nl >> 12;
    int lq = nl & 4095;

    float a[POINTS], t[POINTS];
    float mx = -1e30f;
#pragma unroll
    for (int p = 0; p < POINTS; ++p) {
        a[p] = aw_raw[(size_t)nl * 32 + h * POINTS + p];
        mx = fmaxf(mx, a[p]);
    }
    float s = 0.0f;
#pragma unroll
    for (int p = 0; p < POINTS; ++p) { a[p] = expf(a[p] - mx); s += a[p]; }
    float inv = 1.0f / s;
#pragma unroll
    for (int p = 0; p < POINTS; ++p) {
        float acc = 0.0f;
#pragma unroll
        for (int hin = 0; hin < HEADS; ++hin)
            acc += bias8[((size_t)((n * HEADS + hin) * LQ + lq) * POINTS + p) * 8 + h];
        t[p] = a[p] * inv + tanhf(acc * 0.125f);
    }
    mx = -1e30f;
#pragma unroll
    for (int p = 0; p < POINTS; ++p) mx = fmaxf(mx, t[p]);
    s = 0.0f;
#pragma unroll
    for (int p = 0; p < POINTS; ++p) { t[p] = expf(t[p] - mx); s += t[p]; }
    inv = 1.0f / s;
#pragma unroll
    for (int p = 0; p < POINTS; ++p) {
        float v = t[p] * inv;
        attn[((size_t)nl * HEADS + h) * POINTS + p] = v;
        attn_out[((size_t)(n * HEADS + h) * LQ + lq) * POINTS + p] = v;  // (N,h,1,Lq,P)
    }
}

// ---------------- bilinear sampling + attention reduce (wave per (n,lq,h)) ----
__global__ void sample_kernel(const float* __restrict__ value,
                              const float* __restrict__ loc,
                              const float* __restrict__ attn,
                              __bf16* __restrict__ outpre) {
    const int lane = threadIdx.x;                       // channel d
    int w = blockIdx.x * blockDim.y + threadIdx.y;      // (n*LQ+lq)*HEADS + h
    if (w >= NLQ * HEADS) return;
    int h  = w & 7;
    int nl = w >> 3;
    int n  = nl >> 12;
    const float* vb = value + (size_t)(n * HEADS + h) * HW * DH;
    float acc = 0.0f;
#pragma unroll
    for (int p = 0; p < POINTS; ++p) {
        int li = ((nl * HEADS + h) * POINTS + p) * 2;
        float gx = fminf(fmaxf(loc[li],     -1.0f), 1.0f);
        float gy = fminf(fmaxf(loc[li + 1], -1.0f), 1.0f);
        float ap = attn[(size_t)(nl * HEADS + h) * POINTS + p];
        float x = (gx + 1.0f) * 32.0f - 0.5f;   // ((gx+1)*W - 1) * 0.5, W=64
        float y = (gy + 1.0f) * 32.0f - 0.5f;
        float x0 = floorf(x), y0 = floorf(y);
        float wx1 = x - x0, wy1 = y - y0;
        float wx0 = 1.0f - wx1, wy0 = 1.0f - wy1;
        int ix0 = (int)x0, iy0 = (int)y0;
        float sv = 0.0f;
#pragma unroll
        for (int cy = 0; cy < 2; ++cy) {
            int yy = iy0 + cy;
            if (yy < 0 || yy >= HH) continue;
            float wy = cy ? wy1 : wy0;
#pragma unroll
            for (int cx = 0; cx < 2; ++cx) {
                int xx = ix0 + cx;
                if (xx < 0 || xx >= WW) continue;
                float wx = cx ? wx1 : wx0;
                sv += wx * wy * vb[(size_t)(yy * WW + xx) * DH + lane];
            }
        }
        acc += ap * sv;
    }
    outpre[(size_t)nl * C_DIM + h * DH + lane] = (__bf16)acc;  // bf16 A for out-proj
}

// ---------------- launch ----------------
extern "C" void kernel_launch(void* const* d_in, const int* in_sizes, int n_in,
                              void* d_out, int out_size, void* d_ws, size_t ws_size,
                              hipStream_t stream) {
    const float* query  = (const float*)d_in[0];
    const float* refp   = (const float*)d_in[1];
    const float* inpf   = (const float*)d_in[2];
    const float* W_off  = (const float*)d_in[3];
    const float* b_off  = (const float*)d_in[4];
    const float* W_attn = (const float*)d_in[5];
    const float* b_attn = (const float*)d_in[6];
    const float* W_v    = (const float*)d_in[7];
    const float* b_v    = (const float*)d_in[8];
    const float* W_o    = (const float*)d_in[9];
    const float* b_o    = (const float*)d_in[10];
    const float* cW1    = (const float*)d_in[11];
    const float* cb1    = (const float*)d_in[12];
    const float* cW2    = (const float*)d_in[13];
    const float* cb2    = (const float*)d_in[14];
    const float* cW3    = (const float*)d_in[15];
    const float* cb3    = (const float*)d_in[16];

    char* ws = (char*)d_ws;
    size_t o = 0;
    auto take = [&](size_t bytes) { size_t r = o; o += (bytes + 255) & ~(size_t)255; return r; };
    __bf16* qbf      = (__bf16*)(ws + take((size_t)NLQ * C_DIM * 2));
    __bf16* ifbf     = (__bf16*)(ws + take((size_t)NLQ * C_DIM * 2));
    __bf16* outprebf = (__bf16*)(ws + take((size_t)NLQ * C_DIM * 2));
    float*  value_ws = (float*) (ws + take((size_t)N_B * HEADS * HW * DH * 4));
    float*  offsraw  = (float*) (ws + take((size_t)NLQ * 64 * 4));
    float*  awraw    = (float*) (ws + take((size_t)NLQ * 32 * 4));
    float*  loc_ws   = (float*) (ws + take((size_t)NLQ * HEADS * POINTS * 2 * 4));
    float*  bias8    = (float*) (ws + take((size_t)ROWSCPB * 8 * 4));
    float*  attn_ws  = (float*) (ws + take((size_t)NLQ * HEADS * POINTS * 4));
    __bf16* wvp      = (__bf16*)(ws + take((size_t)8 * 16 * 512 * 2));
    __bf16* woffp    = (__bf16*)(ws + take((size_t)8 * 4 * 512 * 2));
    __bf16* wattnp   = (__bf16*)(ws + take((size_t)8 * 2 * 512 * 2));
    __bf16* w2p      = (__bf16*)(ws + take((size_t)2 * 4 * 512 * 2));
    __bf16* w3p      = (__bf16*)(ws + take((size_t)2 * 1 * 512 * 2));
    __bf16* wop      = (__bf16*)(ws + take((size_t)8 * 16 * 512 * 2));

    float* out_main = (float*)d_out;                         // (N, Lq, C)
    float* out_attn = (float*)d_out + (size_t)NLQ * C_DIM;   // (N, heads, 1, Lq, P)

    // 1) bf16 conversions
    f32_to_bf16_kernel<<<(NLQ * C_DIM) / 256, 256, 0, stream>>>(query, qbf, NLQ * C_DIM);
    f32_to_bf16_kernel<<<(NLQ * C_DIM) / 256, 256, 0, stream>>>(inpf, ifbf, NLQ * C_DIM);

    // 2) weight packing into WMMA-B fragment layout
    pack_b_kernel<<<(8 * 16 * 512) / 256, 256, 0, stream>>>(W_v,    wvp,    256, 256, 256);
    pack_b_kernel<<<(8 * 4  * 512) / 256, 256, 0, stream>>>(W_off,  woffp,  256, 64,  64);
    pack_b_kernel<<<(8 * 2  * 512) / 256, 256, 0, stream>>>(W_attn, wattnp, 256, 32,  32);
    pack_b_kernel<<<(2 * 4  * 512) / 256, 256, 0, stream>>>(cW2,    w2p,    64,  64,  64);
    pack_b_kernel<<<(2 * 1  * 512) / 256, 256, 0, stream>>>(cW3,    w3p,    64,  8,   16);
    pack_b_kernel<<<(8 * 16 * 512) / 256, 256, 0, stream>>>(W_o,    wop,    256, 256, 256);

    dim3 wblk(32, 8);
    // 3) value projection GEMM, scattered to (n*8+h, m, d)
    wmma_gemm_kernel<1, 16><<<2048, wblk, 0, stream>>>(ifbf, wvp, b_v, value_ws,
                                                       NLQ, 0);
    // 4) offset / attention-weight projections
    wmma_gemm_kernel<0, 4><<<512, wblk, 0, stream>>>(qbf, woffp, b_off, offsraw,
                                                     NLQ, 64);
    wmma_gemm_kernel<0, 2><<<256, wblk, 0, stream>>>(qbf, wattnp, b_attn, awraw,
                                                     NLQ, 32);
    // 5) sampling locations
    loc_kernel<<<(NLQ * HEADS * POINTS) / 256, 256, 0, stream>>>(offsraw, refp, loc_ws);
    // 6) fused CPB MLP (WMMA layers 2 & 3)
    cpb_kernel<<<ROWSCPB / (16 * 8), wblk, 0, stream>>>(loc_ws, refp, cW1, cb1,
                                                        w2p, cb2, w3p, cb3, bias8);
    // 7) attention weights (double softmax + tanh bias) + attn output
    attn_kernel<<<(NLQ * HEADS) / 256, 256, 0, stream>>>(awraw, bias8, attn_ws, out_attn);
    // 8) bilinear sampling + reduce over points -> bf16 A matrix
    sample_kernel<<<(NLQ * HEADS) / 8, wblk, 0, stream>>>(value_ws, loc_ws, attn_ws, outprebf);
    // 9) output projection GEMM -> d_out
    wmma_gemm_kernel<0, 16><<<2048, wblk, 0, stream>>>(outprebf, wop, b_o, out_main,
                                                       NLQ, 256);
    (void)in_sizes; (void)n_in; (void)out_size; (void)ws_size;
}
